// QuantLinear_15530601742354
// MI455X (gfx1250) — matrix operand order repaired and checked by reading, not compiled, not run on hardware
//
#include <hip/hip_runtime.h>

// ---- CDNA5 WMMA vector types ----
typedef __attribute__((ext_vector_type(16))) __bf16          v16bf;
typedef __attribute__((ext_vector_type(8)))  float           v8f;
typedef __attribute__((ext_vector_type(4)))  unsigned short  ushort4v;
typedef __attribute__((ext_vector_type(8)))  unsigned short  ushort8v;
typedef __attribute__((ext_vector_type(16))) unsigned short  ushort16v;

#define BM 256
#define BN 128
#define BK 64
#define LDT 72        // tile k-stride in bf16 elements: BK + 8 pad (144B rows = 16B*9)
#define NTHREADS 512  // 16 wave32: 4 (M) x 4 (N) wave grid, 64x32 per wave

__device__ __forceinline__ unsigned short f32_to_bf16_bits(float f) {
  __bf16 h = (__bf16)f;  // RNE; lowers to v_cvt_pk_bf16_f32 on gfx1250
  return __builtin_bit_cast(unsigned short, h);
}

__global__ __launch_bounds__(NTHREADS)
void qlinear_wmma_bf16(const float* __restrict__ x,
                       const int*   __restrict__ qweight,
                       const int*   __restrict__ sign_map,
                       const float* __restrict__ scales,
                       float*       __restrict__ out,
                       int M, int K, int N) {
  __shared__ __align__(16) unsigned short Atile[BM * LDT]; // [row][k] bf16
  __shared__ __align__(16) unsigned short Btile[BN * LDT]; // [col][k] bf16 (W^T tile)

  const int tid     = threadIdx.x;
  const int rowBase = blockIdx.y * BM;
  const int colBase = blockIdx.x * BN;

  const int wave  = tid >> 5;
  const int lane  = tid & 31;
  const int wm    = wave >> 2;   // 0..3 -> 64 rows each
  const int wn    = wave & 3;    // 0..3 -> 32 cols each
  const int laneM = lane & 15;
  const int hi    = lane >> 4;   // lane half -> K subrange per ISA frag layout

  // dequant mapping: thread -> (column, 16-k slice)
  const int colL = tid & (BN - 1);
  const int part = tid >> 7;     // 0..3
  const int colG = colBase + colL;
  const float scale = scales[colG];

  // ---- per-thread pre-scaled bf16 product tables (scale is loop-invariant) ----
  // prod[c] = bf16_bits(INVERSE_TABLE[c] * scale); split into lo/hi byte tables.
  unsigned short pb[16];
#pragma unroll
  for (int c = 0; c < 16; ++c) {
    const float fc  = (float)c;
    const float mag = (c < 8) ? (0.5f * fc) : ((c < 13) ? (fc - 4.0f) : (2.0f * fc - 16.0f));
    pb[c] = f32_to_bf16_bits(mag * scale);
  }
#define PK_LO(a) ((unsigned)(pb[(a)] & 0xFF) | ((unsigned)(pb[(a)+1] & 0xFF) << 8) | \
                  ((unsigned)(pb[(a)+2] & 0xFF) << 16) | ((unsigned)(pb[(a)+3] & 0xFF) << 24))
#define PK_HI(a) ((unsigned)(pb[(a)] >> 8) | ((unsigned)(pb[(a)+1] >> 8) << 8) | \
                  ((unsigned)(pb[(a)+2] >> 8) << 16) | ((unsigned)(pb[(a)+3] >> 8) << 24))
  const unsigned TL0 = PK_LO(0), TL1 = PK_LO(4), TL2 = PK_LO(8),  TL3 = PK_LO(12);
  const unsigned TH0 = PK_HI(0), TH1 = PK_HI(4), TH2 = PK_HI(8),  TH3 = PK_HI(12);
#undef PK_LO
#undef PK_HI

  v8f acc[4][2];
#pragma unroll
  for (int mi = 0; mi < 4; ++mi)
#pragma unroll
    for (int ni = 0; ni < 2; ++ni)
      acc[mi][ni] = (v8f){0.f, 0.f, 0.f, 0.f, 0.f, 0.f, 0.f, 0.f};

  const int kTiles = K / BK;

  // ---- software pipeline registers (next tile staged in VGPRs) ----
  float4   ax[8];
  unsigned bw[2];
  unsigned bsign;

  auto stage_tile = [&](int kb) {
#pragma unroll
    for (int j = 0; j < 8; ++j) {
      const int i  = tid + NTHREADS * j;   // 0..4095 float4 slots
      const int r  = i >> 4;               // 16 float4 per 64-wide row
      const int c4 = (i & 15) << 2;
      ax[j] = *(const float4*)(x + (size_t)(rowBase + r) * K + (size_t)kb * BK + c4);
    }
    bsign = (unsigned)sign_map[(size_t)(kb * 2 + (part >> 1)) * N + colG];
#pragma unroll
    for (int q = 0; q < 2; ++q)
      bw[q] = (unsigned)qweight[(size_t)(kb * 8 + part * 2 + q) * N + colG];
  };

  stage_tile(0);

  for (int kb = 0; kb < kTiles; ++kb) {
    // ---- A: staged fp32 regs -> bf16 LDS (v_cvt_pk_bf16_f32) ----
#pragma unroll
    for (int j = 0; j < 8; ++j) {
      const int i  = tid + NTHREADS * j;
      const int r  = i >> 4;
      const int c4 = (i & 15) << 2;
      ushort4v p;
      p[0] = f32_to_bf16_bits(ax[j].x);
      p[1] = f32_to_bf16_bits(ax[j].y);
      p[2] = f32_to_bf16_bits(ax[j].z);
      p[3] = f32_to_bf16_bits(ax[j].w);
      *(ushort4v*)&Atile[r * LDT + c4] = p;
    }

    // ---- B: v_perm table-lookup dequant, 8 bf16 weights per qword ----
#pragma unroll
    for (int q = 0; q < 2; ++q) {
      const unsigned w  = bw[q];
      const unsigned ce = w & 0x0F0F0F0Fu;          // codes of k = 0,2,4,6 (bytes)
      const unsigned co = (w >> 4) & 0x0F0F0F0Fu;   // codes of k = 1,3,5,7

      // even group: two table-half perms merged by code bit3 byte-mask
      const unsigned c7e = ce & 0x07070707u;
      const unsigned m8e = ((ce >> 3) & 0x01010101u) * 0xFFu;  // 0xFF where code>=8
      const unsigned loE = (__builtin_amdgcn_perm(TL3, TL2, c7e) & m8e) |
                           (__builtin_amdgcn_perm(TL1, TL0, c7e) & ~m8e);
      const unsigned hiE = (__builtin_amdgcn_perm(TH3, TH2, c7e) & m8e) |
                           (__builtin_amdgcn_perm(TH1, TH0, c7e) & ~m8e);
      // odd group
      const unsigned c7o = co & 0x07070707u;
      const unsigned m8o = ((co >> 3) & 0x01010101u) * 0xFFu;
      const unsigned loO = (__builtin_amdgcn_perm(TL3, TL2, c7o) & m8o) |
                           (__builtin_amdgcn_perm(TL1, TL0, c7o) & ~m8o);
      const unsigned hiO = (__builtin_amdgcn_perm(TH3, TH2, c7o) & m8o) |
                           (__builtin_amdgcn_perm(TH1, TH0, c7o) & ~m8o);

      // interleave even/odd -> k-order bytes: P0 = k0..k3, P1 = k4..k7
      const unsigned loP0 = __builtin_amdgcn_perm(loO, loE, 0x05010400u);
      const unsigned loP1 = __builtin_amdgcn_perm(loO, loE, 0x07030602u);
      unsigned       hiP0 = __builtin_amdgcn_perm(hiO, hiE, 0x05010400u);
      unsigned       hiP1 = __builtin_amdgcn_perm(hiO, hiE, 0x07030602u);

      // signs: spread 4 sign bits to byte-top bits (carry-free multiply), XOR into hi bytes
      const unsigned t = bsign >> ((part & 1) * 16 + q * 8);  // bits 0..7 = signs k0..k7
      hiP0 ^= ((t & 0xFu) * 0x10204080u) & 0x80808080u;
      hiP1 ^= (((t >> 4) & 0xFu) * 0x10204080u) & 0x80808080u;

      // assemble 8 bf16 values (lo/hi byte interleave) and store 16B
      uint4 st;
      st.x = __builtin_amdgcn_perm(hiP0, loP0, 0x05010400u);  // bf16(k0)|bf16(k1)<<16
      st.y = __builtin_amdgcn_perm(hiP0, loP0, 0x07030602u);
      st.z = __builtin_amdgcn_perm(hiP1, loP1, 0x05010400u);
      st.w = __builtin_amdgcn_perm(hiP1, loP1, 0x07030602u);
      *(uint4*)&Btile[colL * LDT + part * 16 + q * 8] = st;
    }

    __syncthreads();  // tiles ready

    // stage next k-tile's global loads so HBM latency hides behind WMMAs
    if (kb + 1 < kTiles) {
      stage_tile(kb + 1);
      if (kb + 2 < kTiles) {
        __builtin_prefetch(x + (size_t)(rowBase + (tid >> 4)) * K + (size_t)(kb + 2) * BK, 0, 1);
        __builtin_prefetch(qweight + (size_t)((kb + 2) * 8 + part * 2) * N + colG, 0, 1);
      }
    }

    // ---- 2 k-steps of 32 -> 16 WMMAs per wave ----
#pragma unroll
    for (int ks = 0; ks < 2; ++ks) {
      const int ko = ks * 32;
      v16bf afrag[4];
#pragma unroll
      for (int mi = 0; mi < 4; ++mi) {
        const unsigned short* ap = &Atile[(wm * 64 + mi * 16 + laneM) * LDT + ko];
        const ushort8v lo = *(const ushort8v*)(ap + hi * 8);
        const ushort8v hh = *(const ushort8v*)(ap + hi * 8 + 16);
        ushort16v u;
#pragma unroll
        for (int e = 0; e < 8; ++e) { u[e] = lo[e]; u[e + 8] = hh[e]; }
        afrag[mi] = __builtin_bit_cast(v16bf, u);
      }
      v16bf bfrag[2];
#pragma unroll
      for (int ni = 0; ni < 2; ++ni) {
        const unsigned short* bp = &Btile[(wn * 32 + ni * 16 + laneM) * LDT + ko];
        const ushort8v lo = *(const ushort8v*)(bp + hi * 8);
        const ushort8v hh = *(const ushort8v*)(bp + hi * 8 + 16);
        ushort16v u;
#pragma unroll
        for (int e = 0; e < 8; ++e) { u[e] = lo[e]; u[e + 8] = hh[e]; }
        bfrag[ni] = __builtin_bit_cast(v16bf, u);
      }
#pragma unroll
      for (int mi = 0; mi < 4; ++mi)
#pragma unroll
        for (int ni = 0; ni < 2; ++ni)
          acc[mi][ni] = __builtin_amdgcn_wmma_f32_16x16x32_bf16(
              false, afrag[mi], false, bfrag[ni],
              (short)0, acc[mi][ni], false, false);
    }

    __syncthreads();  // done reading LDS; next iter may overwrite
  }

  // ---- epilogue: C/D layout lane->N, VGPRv->M (+8 for upper lane half);
  //      round accumulator to bf16, widen to f32 (matches reference cast)
#pragma unroll
  for (int mi = 0; mi < 4; ++mi)
#pragma unroll
    for (int ni = 0; ni < 2; ++ni) {
      const int row0 = rowBase + wm * 64 + mi * 16 + hi * 8;
      const int col  = colBase + wn * 32 + ni * 16 + laneM;
      float* op = out + (size_t)row0 * N + col;
#pragma unroll
      for (int v = 0; v < 8; ++v) {
        const unsigned u = ((unsigned)f32_to_bf16_bits(acc[mi][ni][v])) << 16;
        op[(size_t)v * N] = __builtin_bit_cast(float, u);
      }
    }
}

extern "C" void kernel_launch(void* const* d_in, const int* in_sizes, int n_in,
                              void* d_out, int out_size, void* d_ws, size_t ws_size,
                              hipStream_t stream) {
  const float* x        = (const float*)d_in[0];
  const int*   qweight  = (const int*)d_in[1];
  const int*   sign_map = (const int*)d_in[2];
  const float* scales   = (const float*)d_in[3];
  float*       out      = (float*)d_out;

  const long long N  = in_sizes[3];                       // 11008
  const long long IN = ((long long)in_sizes[1] * 8) / N;  // 4096
  const long long M  = (long long)in_sizes[0] / IN;       // 4096

  dim3 grid((unsigned)(N / BN), (unsigned)(M / BM));
  qlinear_wmma_bf16<<<grid, NTHREADS, 0, stream>>>(x, qweight, sign_map, scales, out,
                                                   (int)M, (int)IN, (int)N);
}
